// PointNet_Estimation_53017076302140
// MI455X (gfx1250) — matrix operand order repaired and checked by reading, not compiled, not run on hardware
//
#include <hip/hip_runtime.h>
#include <cstdint>
#include <cstddef>

// ---------------------------------------------------------------------------
// PointNet++ MSG segmentation forward on gfx1250 (MI455X).
// All channel-mixing matmuls use v_wmma_f32_16x16x32_f16 (wave32 WMMA).
// Weights are pre-converted to zero-padded f16 (cin padded to 32, cout to 32),
// activations are kept as zero-padded f16, so every WMMA fragment is exactly
// two unguarded 128-bit loads (ds_load_b128 / global_load_b128).
// SA stages fuse gather + 3xCBR + max-pool with activations in LDS so grouped
// intermediates (up to 268MB if materialized) never touch HBM.
// ---------------------------------------------------------------------------

typedef __attribute__((ext_vector_type(16))) _Float16 v16h;
typedef __attribute__((ext_vector_type(8)))  _Float16 v8h;
typedef __attribute__((ext_vector_type(8)))  float    v8f;

#define BN_EPS 1e-5f

struct CbrP {   // raw fp32 params straight from d_in
  const float* W; const float* b; const float* g; const float* be;
  const float* mn; const float* vr; int cin; int cout;
};
struct CbrF {   // prepped: f16 padded weights + fp32 BN params
  const _Float16* Wh; const float* b; const float* g; const float* be;
  const float* mn; const float* vr; int cinPad; int cout; int coutPad;
};

// ---------------- WMMA fragment loads (16x16x32 f16, wave32) ---------------
// A (16x32): lane m=lane&15, K-half kh=lane>>4. VGPR0-3 hold K = kh*8+0..7,
// VGPR4-7 hold K = 16+kh*8+0..7  -> two contiguous 8xf16 runs, 16 apart.
__device__ __forceinline__ v16h ld_frag_a(const _Float16* p) {
  v8h lo = *(const v8h*)p;
  v8h hi = *(const v8h*)(p + 16);
  return __builtin_shufflevector(lo, hi, 0,1,2,3,4,5,6,7,8,9,10,11,12,13,14,15);
}
// B (32x16): lane n=lane&15 holds K = kh*8+0..7 and 16+kh*8+0..7 of W row n (same as A).
__device__ __forceinline__ v16h ld_frag_b(const _Float16* p) {
  v8h lo = *(const v8h*)p;
  v8h hi = *(const v8h*)(p + 16);
  return __builtin_shufflevector(lo, hi, 0,1,2,3,4,5,6,7,8,9,10,11,12,13,14,15);
}

// One CBR layer for a 16-row LDS tile: f16 in -> WMMA -> BN+ReLU -> f16 out.
__device__ __forceinline__ void cbr_layer_h(const _Float16* in, _Float16* outb,
                                            const CbrF P, int lane) {
  int l15 = lane & 15, kh = (lane >> 4) & 1;
  const _Float16* ap = in + (size_t)l15 * P.cinPad + kh * 8;
  int ntiles = P.coutPad >> 4;
  for (int nt = 0; nt < ntiles; ++nt) {
    const _Float16* wp = P.Wh + (size_t)(nt * 16 + l15) * P.cinPad + kh * 8;
    v8f acc = {};
    for (int kb = 0; kb < P.cinPad; kb += 32) {
      acc = __builtin_amdgcn_wmma_f32_16x16x32_f16(
          false, ld_frag_a(ap + kb), false, ld_frag_b(wp + kb),
          (short)0, acc, false, false);
    }
    int col = nt * 16 + l15;
    float scale = 0.f, shift = 0.f;
    if (col < P.cout) {
      float s = P.g[col] * rsqrtf(P.vr[col] + BN_EPS);
      scale = s;
      shift = (P.b[col] - P.mn[col]) * s + P.be[col];
    }
#pragma unroll
    for (int i = 0; i < 8; ++i) {
      float y = fmaxf(acc[i] * scale + shift, 0.f);  // 0 in pad cols
      outb[(size_t)(kh * 8 + i) * P.coutPad + col] = (_Float16)y;
    }
  }
}

// ---------------- Fused SA branch: gather + 3xCBR + max-pool ---------------
// grid.x = B*np (centroid), grid.y = k/16 (sample tile), block = 1 wave.
// mode 0: msg grouping  g = [points[gidx], xyz[gidx]-new_xyz]
// mode 1: group_all     g = [xyz, points]  (np==1, j = sample index)
#define BUFA_H (16 * 544)
#define BUFB_H (16 * 1024)

__global__ void __launch_bounds__(32) sa_fused(
    const float* xyz, const float* srcp, int N, int Cpts,
    const float* new_xyz, const int* gidx,
    int np, int k, int mode,
    CbrF L0, CbrF L1, CbrF L2,
    float* out, int CoutTotal, int outOff)
{
  __shared__ __align__(16) _Float16 bufA[BUFA_H];
  __shared__ __align__(16) _Float16 bufB[BUFB_H];
  int lane = threadIdx.x;
  int cid  = blockIdx.x;      // b*np + p
  int tile = blockIdx.y;
  int b = cid / np;
  int cin0 = (mode == 0) ? (Cpts + 3) : (Cpts + 3);
  int cin0Pad = L0.cinPad;

  // gather 16 samples x cin0Pad channels into LDS (zero-padded f16)
  for (int t = lane; t < 16 * cin0Pad; t += 32) {
    int s = t / cin0Pad, c = t % cin0Pad;
    int samp = tile * 16 + s;
    float v = 0.f;
    if (c < cin0) {
      if (mode == 0) {
        int j = gidx[(size_t)cid * k + samp];
        if (c < Cpts) v = srcp[((size_t)b * N + j) * Cpts + c];
        else {
          int d = c - Cpts;
          v = xyz[((size_t)b * N + j) * 3 + d] - new_xyz[(size_t)cid * 3 + d];
        }
      } else {
        int j = samp;
        if (c < 3) v = xyz[((size_t)b * N + j) * 3 + c];
        else       v = srcp[((size_t)b * N + j) * Cpts + (c - 3)];
      }
    }
    bufA[s * cin0Pad + c] = (_Float16)v;
  }
  __syncthreads();
  cbr_layer_h(bufA, bufB, L0, lane);
  __syncthreads();
  cbr_layer_h(bufB, bufA, L1, lane);
  __syncthreads();
  cbr_layer_h(bufA, bufB, L2, lane);
  __syncthreads();

  int c2 = L2.cout, ld2 = L2.coutPad;
  for (int c = lane; c < c2; c += 32) {
    float mx = 0.f;  // post-ReLU values are >= 0
    for (int s = 0; s < 16; ++s) mx = fmaxf(mx, (float)bufB[s * ld2 + c]);
    atomicMax((int*)&out[(size_t)cid * CoutTotal + outOff + c], __float_as_int(mx));
  }
}

// ---------------- Plain GEMM + BN + ReLU (f16 in, f16 out) -----------------
// M must be a multiple of 16; X is (M x cinPad) f16; Y is (M x coutPad) f16.
__global__ void __launch_bounds__(32) gemm_cbr_h(
    const _Float16* Xh, CbrF P, _Float16* Yh, int M)
{
  int lane = threadIdx.x;
  int mbase = blockIdx.x * 16, nbase = blockIdx.y * 16;
  int l15 = lane & 15, kh = (lane >> 4) & 1;
  const _Float16* ap = Xh + (size_t)(mbase + l15) * P.cinPad + kh * 8;
  const _Float16* wp = P.Wh + (size_t)(nbase + l15) * P.cinPad + kh * 8;
  v8f acc = {};
  for (int kb = 0; kb < P.cinPad; kb += 32) {
    acc = __builtin_amdgcn_wmma_f32_16x16x32_f16(
        false, ld_frag_a(ap + kb), false, ld_frag_b(wp + kb),
        (short)0, acc, false, false);
  }
  int col = nbase + l15;
  float scale = 0.f, shift = 0.f;
  if (col < P.cout) {
    float s = P.g[col] * rsqrtf(P.vr[col] + BN_EPS);
    scale = s;
    shift = (P.b[col] - P.mn[col]) * s + P.be[col];
  }
#pragma unroll
  for (int i = 0; i < 8; ++i) {
    float y = fmaxf(acc[i] * scale + shift, 0.f);
    Yh[(size_t)(mbase + kh * 8 + i) * P.coutPad + col] = (_Float16)y;
  }
}

// ---------------- GEMM + bias + sigmoid (classifier head, f32 out) ---------
__global__ void __launch_bounds__(32) gemm_sig_h(
    const _Float16* Xh, int cinPad, const _Float16* Wh, const float* bias,
    int cout, float* Yf, int ldY, int M)
{
  int lane = threadIdx.x;
  int mbase = blockIdx.x * 16, nbase = blockIdx.y * 16;
  int l15 = lane & 15, kh = (lane >> 4) & 1;
  const _Float16* ap = Xh + (size_t)(mbase + l15) * cinPad + kh * 8;
  const _Float16* wp = Wh + (size_t)(nbase + l15) * cinPad + kh * 8;
  v8f acc = {};
  for (int kb = 0; kb < cinPad; kb += 32) {
    acc = __builtin_amdgcn_wmma_f32_16x16x32_f16(
        false, ld_frag_a(ap + kb), false, ld_frag_b(wp + kb),
        (short)0, acc, false, false);
  }
  int col = nbase + l15;
  if (col >= cout) return;
  float sh = bias[col];
#pragma unroll
  for (int i = 0; i < 8; ++i) {
    float y = 1.f / (1.f + expf(-(acc[i] + sh)));
    Yf[(size_t)(mbase + kh * 8 + i) * ldY + col] = y;
  }
}

// ---------------- Prep / utility kernels -----------------------------------
__global__ void cvt_w(const float* W, _Float16* Wh, int cin, int cout,
                      int cinPad, int coutPad) {
  int t = blockIdx.x * blockDim.x + threadIdx.x;
  if (t >= coutPad * cinPad) return;
  int r = t / cinPad, c = t % cinPad;
  float v = (r < cout && c < cin) ? W[(size_t)r * cin + c] : 0.f;
  Wh[t] = (_Float16)v;
}

__global__ void zero_f32(float* p, int n) {
  int t = blockIdx.x * blockDim.x + threadIdx.x;
  if (t < n) p[t] = 0.f;
}
__global__ void zero_h(_Float16* p, int n) {
  int t = blockIdx.x * blockDim.x + threadIdx.x;
  if (t < n) p[t] = (_Float16)0.f;
}

// fp32 src (rows x Cs, optional row broadcast) -> f16 dst slab at [*, off..)
__global__ void copy_f2h(const float* src, _Float16* dst, int rows, int Cs,
                         int ldD, int off, int row_div) {
  int t = blockIdx.x * blockDim.x + threadIdx.x;
  if (t >= rows * Cs) return;
  int r = t / Cs, c = t % Cs;
  dst[(size_t)r * ldD + off + c] = (_Float16)src[(size_t)(r / row_div) * Cs + c];
}

__global__ void transpose_xyz(const float* in, float* pts, int B, int N) {
  int t = blockIdx.x * blockDim.x + threadIdx.x;
  if (t >= B * N * 3) return;
  int b = t / (N * 3); int rem = t % (N * 3);
  int n = rem / 3, c = rem % 3;
  pts[((size_t)b * N + n) * 3 + c] = in[(size_t)b * 3 * N + (size_t)c * N + n];
}

// Iterative farthest point sampling; one block per batch.
__global__ void fps_kernel(const float* xyz, int* idx, int N, int np) {
  __shared__ float dist[2048];
  __shared__ float smax[256];
  __shared__ int   sarg[256];
  __shared__ int   sFar;
  int b = blockIdx.x, tid = threadIdx.x;
  for (int j = tid; j < N; j += 256) dist[j] = 1e10f;
  if (tid == 0) sFar = 0;
  __syncthreads();
  for (int i = 0; i < np; ++i) {
    int far = sFar;
    if (tid == 0) idx[b * np + i] = far;
    float cx = xyz[((size_t)b * N + far) * 3 + 0];
    float cy = xyz[((size_t)b * N + far) * 3 + 1];
    float cz = xyz[((size_t)b * N + far) * 3 + 2];
    float best = -1.f; int barg = 0;
    for (int j = tid; j < N; j += 256) {
      float dx = xyz[((size_t)b * N + j) * 3 + 0] - cx;
      float dy = xyz[((size_t)b * N + j) * 3 + 1] - cy;
      float dz = xyz[((size_t)b * N + j) * 3 + 2] - cz;
      float d = dx * dx + dy * dy + dz * dz;
      float nd = fminf(dist[j], d);
      dist[j] = nd;
      if (nd > best) { best = nd; barg = j; }
    }
    smax[tid] = best; sarg[tid] = barg;
    __syncthreads();
    for (int sft = 128; sft > 0; sft >>= 1) {
      if (tid < sft) {
        if (smax[tid + sft] > smax[tid] ||
            (smax[tid + sft] == smax[tid] && sarg[tid + sft] < sarg[tid])) {
          smax[tid] = smax[tid + sft]; sarg[tid] = sarg[tid + sft];
        }
      }
      __syncthreads();
    }
    if (tid == 0) sFar = sarg[0];
    __syncthreads();
  }
}

__global__ void gather_xyz(const float* xyz, const int* idx, float* nxyz,
                           int N, int np, int B) {
  int t = blockIdx.x * blockDim.x + threadIdx.x;
  if (t >= B * np) return;
  int b = t / np;
  int j = idx[t];
  for (int c = 0; c < 3; ++c)
    nxyz[(size_t)t * 3 + c] = xyz[((size_t)b * N + j) * 3 + c];
}

// Ball query: first k indices (ascending) within radius, padded with first hit.
__global__ void ball_query(const float* xyz, const float* nxyz, int* gidx,
                           int B, int N, int np, int k, float r2) {
  int t = blockIdx.x * blockDim.x + threadIdx.x;
  if (t >= B * np) return;
  int b = t / np;
  float cx = nxyz[(size_t)t * 3], cy = nxyz[(size_t)t * 3 + 1], cz = nxyz[(size_t)t * 3 + 2];
  int* g = gidx + (size_t)t * k;
  int cnt = 0, first = 0; bool has = false;
  for (int j = 0; j < N && cnt < k; ++j) {
    float dx = xyz[((size_t)b * N + j) * 3] - cx;
    float dy = xyz[((size_t)b * N + j) * 3 + 1] - cy;
    float dz = xyz[((size_t)b * N + j) * 3 + 2] - cz;
    if (dx * dx + dy * dy + dz * dz <= r2) {
      if (!has) { first = j; has = true; }
      g[cnt++] = j;
    }
  }
  for (; cnt < k; ++cnt) g[cnt] = first;
}

// 3-NN inverse-distance interpolation, f16 features in, f16 slab out.
__global__ void three_interp_h(const float* xyz1, const float* xyz2,
                               const _Float16* p2, _Float16* dst,
                               int B, int N, int M2, int C, int ldP, int ldD, int off) {
  int t = blockIdx.x * blockDim.x + threadIdx.x;
  if (t >= B * N) return;
  int b = t / N;
  float px = xyz1[(size_t)t * 3], py = xyz1[(size_t)t * 3 + 1], pz = xyz1[(size_t)t * 3 + 2];
  float d0 = 1e30f, d1 = 1e30f, d2 = 1e30f; int i0 = 0, i1 = 0, i2 = 0;
  for (int j = 0; j < M2; ++j) {
    float dx = xyz2[((size_t)b * M2 + j) * 3] - px;
    float dy = xyz2[((size_t)b * M2 + j) * 3 + 1] - py;
    float dz = xyz2[((size_t)b * M2 + j) * 3 + 2] - pz;
    float d = dx * dx + dy * dy + dz * dz;
    if (d < d0)      { d2 = d1; i2 = i1; d1 = d0; i1 = i0; d0 = d; i0 = j; }
    else if (d < d1) { d2 = d1; i2 = i1; d1 = d;  i1 = j; }
    else if (d < d2) { d2 = d;  i2 = j; }
  }
  float w0 = 1.f / (d0 + 1e-8f), w1 = 1.f / (d1 + 1e-8f), w2 = 1.f / (d2 + 1e-8f);
  float ws = w0 + w1 + w2; w0 /= ws; w1 /= ws; w2 /= ws;
  const _Float16* r0 = p2 + ((size_t)b * M2 + i0) * ldP;
  const _Float16* r1 = p2 + ((size_t)b * M2 + i1) * ldP;
  const _Float16* r2p = p2 + ((size_t)b * M2 + i2) * ldP;
  _Float16* o = dst + (size_t)t * ldD + off;
  for (int c = 0; c < C; ++c)
    o[c] = (_Float16)(w0 * (float)r0[c] + w1 * (float)r1[c] + w2 * (float)r2p[c]);
}

// Fold per-class head: Wfold[k,c] = sum_d W1[k,d,c]*s[k,d]*W2[k,d]
__global__ void cls_fold_w(const float* W1, const float* gamma, const float* var,
                           const float* W2, float* wf, int K, int D) {
  int t = blockIdx.x * blockDim.x + threadIdx.x;
  if (t >= K * D) return;
  int cls = t / D, c = t % D;
  const float* w1 = W1 + (size_t)cls * D * D;
  float acc = 0.f;
  for (int d = 0; d < D; ++d) {
    float s = gamma[cls * D + d] * rsqrtf(var[cls * D + d] + BN_EPS);
    acc += w1[(size_t)d * D + c] * s * W2[cls * D + d];
  }
  wf[cls * D + c] = acc;
}

__global__ void cls_fold_c(const float* b1, const float* mean, const float* var,
                           const float* gamma, const float* beta, const float* W2,
                           const float* b2, float* cf, int K, int D) {
  int cls = blockIdx.x * blockDim.x + threadIdx.x;
  if (cls >= K) return;
  float acc = b2[cls];
  for (int d = 0; d < D; ++d) {
    int i = cls * D + d;
    float s = gamma[i] * rsqrtf(var[i] + BN_EPS);
    acc += ((b1[i] - mean[i]) * s + beta[i]) * W2[i];
  }
  cf[cls] = acc;
}

// ---------------------------------------------------------------------------

static inline CbrP take_cbr(void* const* d_in, int& i, int cin, int cout) {
  CbrP p;
  p.W  = (const float*)d_in[i++]; p.b  = (const float*)d_in[i++];
  p.g  = (const float*)d_in[i++]; p.be = (const float*)d_in[i++];
  p.mn = (const float*)d_in[i++]; p.vr = (const float*)d_in[i++];
  p.cin = cin; p.cout = cout;
  return p;
}

extern "C" void kernel_launch(void* const* d_in, const int* in_sizes, int n_in,
                              void* d_out, int out_size, void* d_ws, size_t ws_size,
                              hipStream_t stream) {
  (void)in_sizes; (void)n_in; (void)out_size; (void)ws_size;
  const int B = 8, N0 = 2048, NP1 = 512, NP2 = 128, K = 18, D = 128;
  const int TB = 256;

  // ---- unpack params (recursive insertion-order flattening) ----
  int ii = 0;
  const float* xyz_in = (const float*)d_in[ii++];
  CbrP sa1b[3][3];
  { int cfg[3][4] = {{6,32,32,64},{6,64,64,128},{6,64,96,128}};
    for (int br = 0; br < 3; ++br)
      for (int l = 0; l < 3; ++l) sa1b[br][l] = take_cbr(d_in, ii, cfg[br][l], cfg[br][l+1]); }
  CbrP sa2b[2][3];
  { int cfg[2][4] = {{323,128,128,256},{323,128,196,256}};
    for (int br = 0; br < 2; ++br)
      for (int l = 0; l < 3; ++l) sa2b[br][l] = take_cbr(d_in, ii, cfg[br][l], cfg[br][l+1]); }
  CbrP sa3l[3];
  { int cfg[4] = {515,256,512,1024};
    for (int l = 0; l < 3; ++l) sa3l[l] = take_cbr(d_in, ii, cfg[l], cfg[l+1]); }
  CbrP fp3l[2];
  { int cfg[3] = {1536,256,256};
    for (int l = 0; l < 2; ++l) fp3l[l] = take_cbr(d_in, ii, cfg[l], cfg[l+1]); }
  CbrP fp2l[2];
  { int cfg[3] = {576,256,128};
    for (int l = 0; l < 2; ++l) fp2l[l] = take_cbr(d_in, ii, cfg[l], cfg[l+1]); }
  CbrP fp1l[2];
  { int cfg[3] = {134,128,128};
    for (int l = 0; l < 2; ++l) fp1l[l] = take_cbr(d_in, ii, cfg[l], cfg[l+1]); }
  const float* cW1 = (const float*)d_in[ii++];
  const float* cB1 = (const float*)d_in[ii++];
  const float* cGa = (const float*)d_in[ii++];
  const float* cBe = (const float*)d_in[ii++];
  const float* cMn = (const float*)d_in[ii++];
  const float* cVr = (const float*)d_in[ii++];
  const float* cW2 = (const float*)d_in[ii++];
  const float* cB2 = (const float*)d_in[ii++];

  // ---- workspace bump allocator ----
  char* wsBase = (char*)d_ws;
  size_t wsOff = 0;
  auto walloc = [&](size_t bytes) -> void* {
    wsOff = (wsOff + 255) & ~(size_t)255;
    void* p = wsBase + wsOff;
    wsOff += bytes;
    return p;
  };
  float* pts     = (float*)walloc((size_t)B * N0 * 3 * 4);
  int*   idx1    = (int*)  walloc((size_t)B * NP1 * 4);
  float* nx1     = (float*)walloc((size_t)B * NP1 * 3 * 4);
  int*   idx2    = (int*)  walloc((size_t)B * NP2 * 4);
  float* nx2     = (float*)walloc((size_t)B * NP2 * 3 * 4);
  int*   gidx    = (int*)  walloc((size_t)B * NP1 * 128 * 4);
  float* l1p     = (float*)walloc((size_t)B * NP1 * 320 * 4);
  float* l2p     = (float*)walloc((size_t)B * NP2 * 512 * 4);
  float* l3p     = (float*)walloc((size_t)B * 1024 * 4);
  float* wfold   = (float*)walloc((size_t)K * D * 4);
  float* cfold   = (float*)walloc((size_t)K * 4);
  _Float16* whPool  = (_Float16*)walloc((size_t)8 * 1024 * 1024);     // f16 weights
  _Float16* fp3in_h = (_Float16*)walloc((size_t)B * NP2 * 1536 * 2);
  _Float16* fp2in_h = (_Float16*)walloc((size_t)B * NP1 * 576 * 2);
  _Float16* fp1in_h = (_Float16*)walloc((size_t)B * N0 * 160 * 2);
  _Float16* mid_h   = (_Float16*)walloc((size_t)B * N0 * 128 * 2);
  _Float16* l2pf_h  = (_Float16*)walloc((size_t)B * NP2 * 256 * 2);
  _Float16* l1pf_h  = (_Float16*)walloc((size_t)B * NP1 * 128 * 2);
  _Float16* l0f_h   = (_Float16*)walloc((size_t)B * N0 * 128 * 2);

  auto grid1 = [&](int n) { return dim3((n + TB - 1) / TB); };
  auto pad32 = [](int x) { return (x + 31) & ~31; };

  // ---- pre-convert all weights to padded f16 ----
  size_t whOff = 0;
  auto prep = [&](const CbrP& p) -> CbrF {
    CbrF f;
    f.cinPad = pad32(p.cin); f.cout = p.cout; f.coutPad = pad32(p.cout);
    _Float16* dst = whPool + whOff;
    whOff += (size_t)f.coutPad * f.cinPad;
    cvt_w<<<grid1(f.coutPad * f.cinPad), TB, 0, stream>>>(p.W, dst, p.cin, p.cout,
                                                          f.cinPad, f.coutPad);
    f.Wh = dst;
    f.b = p.b; f.g = p.g; f.be = p.be; f.mn = p.mn; f.vr = p.vr;
    return f;
  };
  CbrF sa1f[3][3], sa2f[2][3], sa3f[3], fp3f[2], fp2f[2], fp1f[2];
  for (int br = 0; br < 3; ++br) for (int l = 0; l < 3; ++l) sa1f[br][l] = prep(sa1b[br][l]);
  for (int br = 0; br < 2; ++br) for (int l = 0; l < 3; ++l) sa2f[br][l] = prep(sa2b[br][l]);
  for (int l = 0; l < 3; ++l) sa3f[l] = prep(sa3l[l]);
  for (int l = 0; l < 2; ++l) fp3f[l] = prep(fp3l[l]);
  for (int l = 0; l < 2; ++l) fp2f[l] = prep(fp2l[l]);
  for (int l = 0; l < 2; ++l) fp1f[l] = prep(fp1l[l]);
  _Float16* wfoldh = whPool + whOff;  whOff += (size_t)32 * 128;

  auto launch_gemm = [&](const _Float16* Xh, int M, const CbrF& P, _Float16* Yh) {
    dim3 g(M / 16, P.coutPad / 16);
    gemm_cbr_h<<<g, 32, 0, stream>>>(Xh, P, Yh, M);
  };

  // ---- stage 0: transpose + zero atomic-max targets & padded concat buf ----
  transpose_xyz<<<grid1(B * N0 * 3), TB, 0, stream>>>(xyz_in, pts, B, N0);
  zero_f32<<<grid1(B * NP1 * 320), TB, 0, stream>>>(l1p, B * NP1 * 320);
  zero_f32<<<grid1(B * NP2 * 512), TB, 0, stream>>>(l2p, B * NP2 * 512);
  zero_f32<<<grid1(B * 1024), TB, 0, stream>>>(l3p, B * 1024);
  zero_h<<<grid1(B * N0 * 160), TB, 0, stream>>>(fp1in_h, B * N0 * 160);

  // ---- SA1 (MSG, np=512) ----
  fps_kernel<<<B, 256, 0, stream>>>(pts, idx1, N0, NP1);
  gather_xyz<<<grid1(B * NP1), TB, 0, stream>>>(pts, idx1, nx1, N0, NP1, B);
  { const float rr[3] = {0.1f, 0.2f, 0.4f}; const int kk[3] = {32, 64, 128};
    const int oo[3] = {0, 64, 192};
    for (int br = 0; br < 3; ++br) {
      ball_query<<<grid1(B * NP1), TB, 0, stream>>>(pts, nx1, gidx, B, N0, NP1,
                                                    kk[br], rr[br] * rr[br]);
      dim3 g(B * NP1, kk[br] / 16);
      sa_fused<<<g, 32, 0, stream>>>(pts, pts, N0, 3, nx1, gidx, NP1, kk[br], 0,
                                     sa1f[br][0], sa1f[br][1], sa1f[br][2],
                                     l1p, 320, oo[br]);
    } }

  // ---- SA2 (MSG, np=128) ----
  fps_kernel<<<B, 256, 0, stream>>>(nx1, idx2, NP1, NP2);
  gather_xyz<<<grid1(B * NP2), TB, 0, stream>>>(nx1, idx2, nx2, NP1, NP2, B);
  { const float rr[2] = {0.4f, 0.8f}; const int kk[2] = {64, 128};
    const int oo[2] = {0, 256};
    for (int br = 0; br < 2; ++br) {
      ball_query<<<grid1(B * NP2), TB, 0, stream>>>(nx1, nx2, gidx, B, NP1, NP2,
                                                    kk[br], rr[br] * rr[br]);
      dim3 g(B * NP2, kk[br] / 16);
      sa_fused<<<g, 32, 0, stream>>>(nx1, l1p, NP1, 320, nx2, gidx, NP2, kk[br], 0,
                                     sa2f[br][0], sa2f[br][1], sa2f[br][2],
                                     l2p, 512, oo[br]);
    } }

  // ---- SA3 (group-all) ----
  { dim3 g(B, NP2 / 16);
    sa_fused<<<g, 32, 0, stream>>>(nx2, l2p, NP2, 512, nullptr, nullptr, 1, NP2, 1,
                                   sa3f[0], sa3f[1], sa3f[2], l3p, 1024, 0); }

  // ---- FP3: concat(l2p, broadcast l3p) -> 256 -> 256 ----
  copy_f2h<<<grid1(B * NP2 * 512), TB, 0, stream>>>(l2p, fp3in_h, B * NP2, 512, 1536, 0, 1);
  copy_f2h<<<grid1(B * NP2 * 1024), TB, 0, stream>>>(l3p, fp3in_h, B * NP2, 1024, 1536, 512, NP2);
  launch_gemm(fp3in_h, B * NP2, fp3f[0], mid_h);
  launch_gemm(mid_h,   B * NP2, fp3f[1], l2pf_h);

  // ---- FP2: interp(l2pf) + l1p -> 256 -> 128 ----
  copy_f2h<<<grid1(B * NP1 * 320), TB, 0, stream>>>(l1p, fp2in_h, B * NP1, 320, 576, 0, 1);
  three_interp_h<<<grid1(B * NP1), TB, 0, stream>>>(nx1, nx2, l2pf_h, fp2in_h,
                                                    B, NP1, NP2, 256, 256, 576, 320);
  launch_gemm(fp2in_h, B * NP1, fp2f[0], mid_h);
  launch_gemm(mid_h,   B * NP1, fp2f[1], l1pf_h);

  // ---- FP1: interp(l1pf) + [xyz,xyz] -> 128 -> 128 ----
  copy_f2h<<<grid1(B * N0 * 3), TB, 0, stream>>>(pts, fp1in_h, B * N0, 3, 160, 0, 1);
  copy_f2h<<<grid1(B * N0 * 3), TB, 0, stream>>>(pts, fp1in_h, B * N0, 3, 160, 3, 1);
  three_interp_h<<<grid1(B * N0), TB, 0, stream>>>(pts, nx1, l1pf_h, fp1in_h,
                                                   B, N0, NP1, 128, 128, 160, 6);
  launch_gemm(fp1in_h, B * N0, fp1f[0], mid_h);
  launch_gemm(mid_h,   B * N0, fp1f[1], l0f_h);

  // ---- classifier head (folded: linear->BN->dot == one GEMM) ----
  cls_fold_w<<<grid1(K * D), TB, 0, stream>>>(cW1, cGa, cVr, cW2, wfold, K, D);
  cls_fold_c<<<1, 32, 0, stream>>>(cB1, cMn, cVr, cGa, cBe, cW2, cB2, cfold, K, D);
  cvt_w<<<grid1(32 * 128), TB, 0, stream>>>(wfold, wfoldh, D, K, D, 32);
  { dim3 g(B * N0 / 16, 2);
    gemm_sig_h<<<g, 32, 0, stream>>>(l0f_h, D, wfoldh, cfold, K,
                                     (float*)d_out, K, B * N0); }
}
